// FunctionBoundaryModel_40922448396299
// MI455X (gfx1250) — matrix operand order repaired and checked by reading, not compiled, not run on hardware
//
#include <hip/hip_runtime.h>
#include <hip/hip_bf16.h>
#include <math.h>

// ---------------- model constants (from reference) ----------------
#define BATCH 2
#define SEQL  2048
#define EMB   256
#define DMODEL 512
#define NLAYER 4
#define NCLS  5
#define DINNER 1024
#define DSTATE 64
#define DRANK  32
#define DCONV  4
#define NTOK  (BATCH*SEQL)        // 4096

typedef __attribute__((ext_vector_type(16))) __bf16 v16bf;
typedef __attribute__((ext_vector_type(8)))  float  v8f;
typedef int v4i __attribute__((vector_size(4 * sizeof(int))));   // matches builtin param

#define AS1 __attribute__((address_space(1)))
#define AS3 __attribute__((address_space(3)))

// probe for the gfx1250 async global->LDS builtins (ASYNCcnt path)
#if defined(__has_builtin)
# if __has_builtin(__builtin_amdgcn_global_load_async_to_lds_b128)
#  define HAVE_ASYNC_LDS 1
# endif
#endif
#ifndef HAVE_ASYNC_LDS
# define HAVE_ASYNC_LDS 0
#endif

// ---------------- helpers ----------------
static __device__ __forceinline__ __bf16 f2bf(float f) {
    union { float f; unsigned u; } v; v.f = f;
    unsigned r = (v.u + 0x7FFFu + ((v.u >> 16) & 1u)) >> 16;
    unsigned short s = (unsigned short)r;
    __bf16 b; __builtin_memcpy(&b, &s, 2);
    return b;
}
static __device__ __forceinline__ float silu(float x) {
    return x * (1.0f / (1.0f + __expf(-x)));
}

// 16B global -> LDS, async (GLOBAL_LOAD_ASYNC_TO_LDS_B128) when available
static __device__ __forceinline__ void async_b128(const __bf16* g, __bf16* l) {
#if HAVE_ASYNC_LDS
    __builtin_amdgcn_global_load_async_to_lds_b128(
        (AS1 v4i*)(v4i*)(__bf16*)g, (AS3 v4i*)(v4i*)l, 0, 0);
#else
    __builtin_memcpy(l, g, 16);
#endif
}
static __device__ __forceinline__ void wait_async0() {
#if HAVE_ASYNC_LDS
# if __has_builtin(__builtin_amdgcn_s_wait_asynccnt)
    __builtin_amdgcn_s_wait_asynccnt(0);
# else
    asm volatile("s_wait_asynccnt 0x0" ::: "memory");
# endif
#endif
}

// ---------------- generic f32 -> bf16 convert ----------------
__global__ void cvt_f32_bf16(const float* __restrict__ src, __bf16* __restrict__ dst, int n) {
    int i = blockIdx.x * blockDim.x + threadIdx.x;
    if (i < n) dst[i] = f2bf(src[i]);
}

// ---------------- embedding + positional ----------------
__global__ void embed_kernel(const int* __restrict__ seq,
                             const float* __restrict__ byte_embed,
                             const float* __restrict__ pos_embed,
                             __bf16* __restrict__ out_bf) {
    int idx = blockIdx.x * blockDim.x + threadIdx.x;      // over NTOK*EMB
    if (idx >= NTOK * EMB) return;
    int tok = idx / EMB, e = idx % EMB;
    int l = tok % SEQL;
    int byte = seq[tok];
    float v = byte_embed[byte * EMB + e] + pos_embed[l * EMB + e];
    out_bf[idx] = f2bf(v);
}

// ---------------- layernorm over D=512, one block per token ----------------
__global__ __launch_bounds__(256)
void ln_kernel(const float* __restrict__ x, const float* __restrict__ g,
               const float* __restrict__ b, __bf16* __restrict__ out_bf) {
    __shared__ float red[256];
    int tok = blockIdx.x, tid = threadIdx.x;
    const float* xr = x + (size_t)tok * DMODEL;
    float v0 = xr[tid], v1 = xr[tid + 256];
    red[tid] = v0 + v1;
    __syncthreads();
    for (int s = 128; s > 0; s >>= 1) { if (tid < s) red[tid] += red[tid + s]; __syncthreads(); }
    float mu = red[0] * (1.0f / DMODEL);
    __syncthreads();
    float d0 = v0 - mu, d1 = v1 - mu;
    red[tid] = d0 * d0 + d1 * d1;
    __syncthreads();
    for (int s = 128; s > 0; s >>= 1) { if (tid < s) red[tid] += red[tid + s]; __syncthreads(); }
    float rstd = rsqrtf(red[0] * (1.0f / DMODEL) + 1e-5f);
    __bf16* ob = out_bf + (size_t)tok * DMODEL;
    ob[tid]       = f2bf(d0 * rstd * g[tid]       + b[tid]);
    ob[tid + 256] = f2bf(d1 * rstd * g[tid + 256] + b[tid + 256]);
}

// ---------------- causal depthwise conv (DC=4) + bias + SiLU ----------------
__global__ void conv_silu_kernel(const float* __restrict__ xr,   // [NTOK, 2*DINNER]
                                 const float* __restrict__ cw,   // [DINNER, DCONV]
                                 const float* __restrict__ cb,   // [DINNER]
                                 float* __restrict__ xs,         // [NTOK, DINNER]
                                 __bf16* __restrict__ xs_bf) {
    int idx = blockIdx.x * blockDim.x + threadIdx.x;              // over NTOK*DINNER
    if (idx >= NTOK * DINNER) return;
    int d = idx % DINNER;
    int tok = idx / DINNER;
    int l = tok % SEQL;
    int b0 = tok - l;                                             // batch row base
    float acc = cb[d];
    #pragma unroll
    for (int j = 0; j < DCONV; ++j) {
        int ll = l - (DCONV - 1) + j;
        if (ll >= 0)
            acc += xr[(size_t)(b0 + ll) * (2 * DINNER) + d] * cw[d * DCONV + j];
    }
    float s = silu(acc);
    xs[idx] = s;
    xs_bf[idx] = f2bf(s);
}

// ---------------- slice dt (first 32 cols of xdbl) to bf16 ----------------
__global__ void dt_slice_kernel(const float* __restrict__ xdbl, __bf16* __restrict__ dtb) {
    int idx = blockIdx.x * blockDim.x + threadIdx.x;              // over NTOK*DRANK
    if (idx >= NTOK * DRANK) return;
    int row = idx / DRANK, c = idx % DRANK;
    dtb[idx] = f2bf(xdbl[(size_t)row * (DRANK + 2 * DSTATE) + c]);
}

// =================================================================
//  bf16 WMMA GEMM:  out[N,M] = A[N,K] @ W[M,K]^T  (+bias, epilogue)
//  block tile 128x128, 8 waves (4 row-groups x 2 col-groups),
//  per wave 32x64 = 8 x v_wmma_f32_16x16x32_bf16 per K-step.
//  Double-buffered LDS; tiles staged with GLOBAL_LOAD_ASYNC_TO_LDS_B128
//  (ASYNCcnt) so the DMA of tile k+1 overlaps the WMMAs of tile k.
//  epi: 0=none, 1=relu, 2=softplus
// =================================================================
__global__ __launch_bounds__(256)
void wmma_gemm(const __bf16* __restrict__ A, const __bf16* __restrict__ W,
               const float* __restrict__ bias, const float* __restrict__ resid,
               float* __restrict__ out, __bf16* __restrict__ out_bf,
               int N, int M, int K, int epi) {
    __shared__ __bf16 As[2][128][40];   // [buf][row][k] padded (80B row stride)
    __shared__ __bf16 Ws[2][128][40];   // [buf][col][k] padded

    const int tid  = threadIdx.x;
    const int lane = tid & 31;
    const int wave = tid >> 5;
    const int rowBase = (wave >> 1) * 32;   // 0,32,64,96
    const int colBase = (wave & 1) * 64;    // 0,64
    const int blkRow = blockIdx.x * 128;
    const int blkCol = blockIdx.y * 128;

    const int lrow  = tid >> 1;             // 0..127
    const int lhalf = tid & 1;              // which 16-element K chunk

    // stage one 128x32 A tile + 128x32 W tile into LDS buffer `buf`
    auto stage = [&](int buf, int k0) {
        {
            const __bf16* src = A + (size_t)(blkRow + lrow) * K + k0 + lhalf * 16;
            __bf16* dst = &As[buf][lrow][lhalf * 16];
            async_b128(src, dst);
            async_b128(src + 8, dst + 8);
        }
        {
            int wcol = blkCol + lrow;
            if (wcol >= M) wcol = M - 1;     // safe garbage; store guard drops it
            const __bf16* src = W + (size_t)wcol * K + k0 + lhalf * 16;
            __bf16* dst = &Ws[buf][lrow][lhalf * 16];
            async_b128(src, dst);
            async_b128(src + 8, dst + 8);
        }
    };

    v8f acc[2][4];
    const v8f vzero = {0,0,0,0,0,0,0,0};
    #pragma unroll
    for (int mi = 0; mi < 2; ++mi)
        #pragma unroll
        for (int ni = 0; ni < 4; ++ni) acc[mi][ni] = vzero;

    // prologue: bring in tile 0
    stage(0, 0);
    wait_async0();
    __syncthreads();

    const int kh = lane >> 4;               // lane K-half group
    const int lm = lane & 15;

    int buf = 0;
    for (int k0 = 0; k0 < K; k0 += 32, buf ^= 1) {
        // issue async DMA for the next tile into the other buffer
        if (k0 + 32 < K) stage(buf ^ 1, k0 + 32);

        // ---- fragments (ISA 16-bit A 16x32 / B 32x16 lane layouts) ----
        union { v16bf v; __bf16 e[16]; } afr[2], bfr[4];
        #pragma unroll
        for (int mi = 0; mi < 2; ++mi) {
            int r = rowBase + mi * 16 + lm;
            __builtin_memcpy(&afr[mi].e[0], &As[buf][r][8 * kh],      16); // K = 8*kh..+7
            __builtin_memcpy(&afr[mi].e[8], &As[buf][r][16 + 8 * kh], 16); // K = 16+8*kh..+7
        }
        #pragma unroll
        for (int ni = 0; ni < 4; ++ni) {
            int c = colBase + ni * 16 + lm;
            __builtin_memcpy(&bfr[ni].e[0], &Ws[buf][c][8 * kh],      16); // K = 8*kh..+7
            __builtin_memcpy(&bfr[ni].e[8], &Ws[buf][c][16 + 8 * kh], 16); // K = 16+8*kh..+7
        }

        #pragma unroll
        for (int mi = 0; mi < 2; ++mi)
            #pragma unroll
            for (int ni = 0; ni < 4; ++ni)
                acc[mi][ni] = __builtin_amdgcn_wmma_f32_16x16x32_bf16(
                    false, afr[mi].v, false, bfr[ni].v,
                    (short)0, acc[mi][ni], false, false);

        // my async copies (into buf^1) must be in LDS before anyone reads them
        wait_async0();
        __syncthreads();
    }

    // ---- epilogue + store (C layout: VGPR i -> M=i / i+8 by lane group, N=lane&15) ----
    const int lg = lane >> 4;
    #pragma unroll
    for (int mi = 0; mi < 2; ++mi) {
        #pragma unroll
        for (int ni = 0; ni < 4; ++ni) {
            int gc = blkCol + colBase + ni * 16 + lm;
            if (gc >= M) continue;
            #pragma unroll
            for (int i = 0; i < 8; ++i) {
                int gr = blkRow + rowBase + mi * 16 + i + 8 * lg;
                float v = acc[mi][ni][i];
                if (bias) v += bias[gc];
                if (epi == 1)      v = fmaxf(v, 0.0f);
                else if (epi == 2) v = (v > 20.0f) ? v : log1pf(__expf(v));
                if (resid) v += resid[(size_t)gr * M + gc];
                out[(size_t)gr * M + gc] = v;
                if (out_bf) out_bf[(size_t)gr * M + gc] = f2bf(v);
            }
        }
    }
}

// =================================================================
//  Selective scan: one wave per (batch, channel); 2 states per lane.
//  Fuses + xs*Dp, * silu(res), bf16 emit for the out-projection GEMM.
// =================================================================
__global__ __launch_bounds__(256)
void scan_kernel(const float* __restrict__ delta,  // [NTOK, DINNER]
                 const float* __restrict__ xs,     // [NTOK, DINNER]
                 const float* __restrict__ xdbl,   // [NTOK, 160]
                 const float* __restrict__ xr,     // [NTOK, 2*DINNER] (res half)
                 const float* __restrict__ A_log,  // [DINNER, DSTATE]
                 const float* __restrict__ Dp,     // [DINNER]
                 __bf16* __restrict__ y_bf) {      // [NTOK, DINNER]
    const int wave = threadIdx.x >> 5;
    const int lane = threadIdx.x & 31;
    const int ch = blockIdx.x * 8 + wave;          // 0 .. BATCH*DINNER-1
    const int b = ch >> 10;                        // / DINNER
    const int d = ch & (DINNER - 1);

    const int s0 = lane, s1 = lane + 32;
    const float a0 = -__expf(A_log[d * DSTATE + s0]);
    const float a1 = -__expf(A_log[d * DSTATE + s1]);
    const float dpv = Dp[d];

    float h0 = 0.0f, h1 = 0.0f;
    const size_t rowBase = (size_t)b * SEQL;
    for (int t = 0; t < SEQL; ++t) {
        const size_t row = rowBase + t;
        float dt = delta[row * DINNER + d];
        float xt = xs[row * DINNER + d];
        const float* xd = xdbl + row * (DRANK + 2 * DSTATE);
        float B0 = xd[DRANK + s0],          B1 = xd[DRANK + s1];
        float C0 = xd[DRANK + DSTATE + s0], C1 = xd[DRANK + DSTATE + s1];
        float du = dt * xt;
        h0 = __expf(dt * a0) * h0 + du * B0;
        h1 = __expf(dt * a1) * h1 + du * B1;
        float p = h0 * C0 + h1 * C1;
        #pragma unroll
        for (int off = 16; off > 0; off >>= 1) p += __shfl_xor(p, off, 32);
        if (lane == 0) {
            float res = xr[row * (2 * DINNER) + DINNER + d];
            float y = (p + xt * dpv) * silu(res);
            y_bf[row * DINNER + d] = f2bf(y);
        }
    }
}

// ---------------- tiny head: [NTOK,128] @ [5,128]^T + b ----------------
__global__ void head3_kernel(const float* __restrict__ h2, const float* __restrict__ w,
                             const float* __restrict__ b, float* __restrict__ out) {
    int idx = blockIdx.x * blockDim.x + threadIdx.x;              // over NTOK*NCLS
    if (idx >= NTOK * NCLS) return;
    int row = idx / NCLS, c = idx % NCLS;
    float acc = b[c];
    const float* hr = h2 + (size_t)row * 128;
    const float* wr = w + c * 128;
    #pragma unroll 8
    for (int k = 0; k < 128; ++k) acc += hr[k] * wr[k];
    out[idx] = acc;
}

// =================================================================
extern "C" void kernel_launch(void* const* d_in, const int* in_sizes, int n_in,
                              void* d_out, int out_size, void* d_ws, size_t ws_size,
                              hipStream_t stream) {
    // inputs (setup_inputs order)
    const int*   seq        = (const int*)  d_in[0];
    const float* byte_embed = (const float*)d_in[1];
    const float* pos_embed  = (const float*)d_in[2];
    const float* in_w       = (const float*)d_in[3];
    const float* in_b       = (const float*)d_in[4];
    const float* ln_g       = (const float*)d_in[5];
    const float* ln_b       = (const float*)d_in[6];
    const float* m_in_w     = (const float*)d_in[7];
    const float* conv_w     = (const float*)d_in[8];
    const float* conv_b     = (const float*)d_in[9];
    const float* xp_w       = (const float*)d_in[10];
    const float* dt_w       = (const float*)d_in[11];
    const float* dt_b       = (const float*)d_in[12];
    const float* A_log      = (const float*)d_in[13];
    const float* Dp         = (const float*)d_in[14];
    const float* m_out_w    = (const float*)d_in[15];
    const float* h1_w       = (const float*)d_in[16];
    const float* h1_b       = (const float*)d_in[17];
    const float* h2_w       = (const float*)d_in[18];
    const float* h2_b       = (const float*)d_in[19];
    const float* h3_w       = (const float*)d_in[20];
    const float* h3_b       = (const float*)d_in[21];
    float* out = (float*)d_out;

    // ---- workspace carve-up ----
    char* ws = (char*)d_ws;
    size_t off = 0;
    auto carve = [&](size_t bytes) -> void* {
        void* p = ws + off; off += (bytes + 255) & ~(size_t)255; return p;
    };
    __bf16* in_w_bf   = (__bf16*)carve((size_t)DMODEL * EMB * 2);
    __bf16* m_in_w_bf = (__bf16*)carve((size_t)NLAYER * 2 * DINNER * DMODEL * 2);
    __bf16* xp_w_bf   = (__bf16*)carve((size_t)NLAYER * (DRANK + 2 * DSTATE) * DINNER * 2);
    __bf16* dt_w_bf   = (__bf16*)carve((size_t)NLAYER * DINNER * DRANK * 2);
    __bf16* m_out_w_bf= (__bf16*)carve((size_t)NLAYER * DMODEL * DINNER * 2);
    __bf16* h1_w_bf   = (__bf16*)carve((size_t)256 * DMODEL * 2);
    __bf16* h2_w_bf   = (__bf16*)carve((size_t)128 * 256 * 2);
    __bf16* Abuf      = (__bf16*)carve((size_t)NTOK * DINNER * 2);
    __bf16* Abuf2     = (__bf16*)carve((size_t)NTOK * 256 * 2);
    __bf16* dtbuf     = (__bf16*)carve((size_t)NTOK * DRANK * 2);
    float*  x    = (float*)carve((size_t)NTOK * DMODEL * 4);
    float*  xr   = (float*)carve((size_t)NTOK * 2 * DINNER * 4);
    float*  xsb  = (float*)carve((size_t)NTOK * DINNER * 4);
    float*  xdbl = (float*)carve((size_t)NTOK * (DRANK + 2 * DSTATE) * 4);
    float*  dlt  = (float*)carve((size_t)NTOK * DINNER * 4);
    float*  hb1  = (float*)carve((size_t)NTOK * 256 * 4);
    float*  hb2  = (float*)carve((size_t)NTOK * 128 * 4);
    (void)ws_size; (void)in_sizes; (void)n_in; (void)out_size;

    auto cvt = [&](const float* s, __bf16* d, int n) {
        cvt_f32_bf16<<<(n + 255) / 256, 256, 0, stream>>>(s, d, n);
    };
    // weight conversions (fp32 -> bf16, once per call)
    cvt(in_w,    in_w_bf,    DMODEL * EMB);
    cvt(m_in_w,  m_in_w_bf,  NLAYER * 2 * DINNER * DMODEL);
    cvt(xp_w,    xp_w_bf,    NLAYER * (DRANK + 2 * DSTATE) * DINNER);
    cvt(dt_w,    dt_w_bf,    NLAYER * DINNER * DRANK);
    cvt(m_out_w, m_out_w_bf, NLAYER * DMODEL * DINNER);
    cvt(h1_w,    h1_w_bf,    256 * DMODEL);
    cvt(h2_w,    h2_w_bf,    128 * 256);

    auto gemm = [&](const __bf16* A, const __bf16* W, const float* bias,
                    const float* resid, float* o, __bf16* obf, int M, int K, int epi) {
        dim3 grid(NTOK / 128, (M + 127) / 128);
        wmma_gemm<<<grid, 256, 0, stream>>>(A, W, bias, resid, o, obf, NTOK, M, K, epi);
    };

    // embed + input projection
    embed_kernel<<<(NTOK * EMB + 255) / 256, 256, 0, stream>>>(seq, byte_embed, pos_embed, Abuf);
    gemm(Abuf, in_w_bf, in_b, nullptr, x, nullptr, DMODEL, EMB, 0);

    for (int i = 0; i < NLAYER; ++i) {
        // layernorm -> bf16
        ln_kernel<<<NTOK, 256, 0, stream>>>(x, ln_g + i * DMODEL, ln_b + i * DMODEL, Abuf);
        // in-projection: [NTOK,512] -> [NTOK,2048]
        gemm(Abuf, m_in_w_bf + (size_t)i * 2 * DINNER * DMODEL, nullptr, nullptr,
             xr, nullptr, 2 * DINNER, DMODEL, 0);
        // causal conv + SiLU
        conv_silu_kernel<<<(NTOK * DINNER + 255) / 256, 256, 0, stream>>>(
            xr, conv_w + (size_t)i * DINNER * DCONV, conv_b + (size_t)i * DINNER, xsb, Abuf);
        // x-projection: [NTOK,1024] -> [NTOK,160]
        gemm(Abuf, xp_w_bf + (size_t)i * (DRANK + 2 * DSTATE) * DINNER, nullptr, nullptr,
             xdbl, nullptr, DRANK + 2 * DSTATE, DINNER, 0);
        // dt low-rank up-projection + softplus: [NTOK,32] -> [NTOK,1024]
        dt_slice_kernel<<<(NTOK * DRANK + 255) / 256, 256, 0, stream>>>(xdbl, dtbuf);
        gemm(dtbuf, dt_w_bf + (size_t)i * DINNER * DRANK, dt_b + (size_t)i * DINNER, nullptr,
             dlt, nullptr, DINNER, DRANK, 2 /*softplus*/);
        // selective scan (fused Dp skip + SiLU gate), emits bf16 y
        scan_kernel<<<(BATCH * DINNER) / 8, 256, 0, stream>>>(
            dlt, xsb, xdbl, xr, A_log + (size_t)i * DINNER * DSTATE, Dp + (size_t)i * DINNER, Abuf);
        // out-projection + residual (in place on x)
        gemm(Abuf, m_out_w_bf + (size_t)i * DMODEL * DINNER, nullptr, x,
             x, nullptr, DMODEL, DINNER, 0);
    }

    // head
    cvt(x, Abuf, NTOK * DMODEL);
    gemm(Abuf,  h1_w_bf, h1_b, nullptr, hb1, Abuf2, 256, DMODEL, 1 /*relu*/);
    gemm(Abuf2, h2_w_bf, h2_b, nullptr, hb2, nullptr, 128, 256, 1 /*relu*/);
    head3_kernel<<<(NTOK * NCLS + 255) / 256, 256, 0, stream>>>(hb2, h3_w, h3_b, out);
}
